// GraphGPSLayer_9577777070225
// MI455X (gfx1250) — compile-verified, matrix operand order checked by reading, not run on hardware
//
#include <hip/hip_runtime.h>

#define N_TOK 2048
#define DMODEL 256
#define NHEAD 8
#define HDIM 32
#define RREL 6
#define NBASE 4
#define NEDGE 65536
#define DFF 1024
#define CATD 512

typedef _Float16 h8   __attribute__((ext_vector_type(8)));
typedef _Float16 v16h __attribute__((ext_vector_type(16)));
typedef float    v8f  __attribute__((ext_vector_type(8)));

union HU { v16h v; h8 h[2]; };

// Load a 16x32 (f16) WMMA A or B operand slice for this lane.
// 16-bit operand striping on gfx1250: lane<16 holds K{0..7} then K{16..23},
// lane>=16 holds K{8..15} then K{24..31}  ->  ko = (lane>=16)*8.
__device__ __forceinline__ v16h load_ab(const _Float16* p, int ko) {
  HU u;
  u.h[0] = *(const h8*)(p + ko);
  u.h[1] = *(const h8*)(p + 16 + ko);
  return u.v;
}

__device__ __forceinline__ float wave_sum32(float v) {
  v += __shfl_xor(v, 1, 32);
  v += __shfl_xor(v, 2, 32);
  v += __shfl_xor(v, 4, 32);
  v += __shfl_xor(v, 8, 32);
  v += __shfl_xor(v, 16, 32);
  return v;
}

// ---------------------------------------------------------------------------
// Generic f16 WMMA GEMM: C[M x Nn] = A[M x K] * Bt[Nn x K]^T (+bias) +epilogue
// One wave computes a 16x64 output tile (4 accumulators).
// MODE: 0 = store f32, 1 = store f16, 2 = GELU->f16, 3 = sigmoid->f32,
//       4 = +bias +add(residual) -> f32
// ---------------------------------------------------------------------------
template <int MODE>
__global__ void __launch_bounds__(128)
gemm_wmma(const _Float16* __restrict__ A, const _Float16* __restrict__ Bt,
          const float* __restrict__ bias, const float* __restrict__ add,
          float* __restrict__ outf, _Float16* __restrict__ outh,
          int M, int Nn, int K) {
  const int lane = threadIdx.x;
  const int grp  = lane >> 4;
  const int ln   = lane & 15;
  const int ko   = grp * 8;
  const int m0   = (blockIdx.x * blockDim.y + threadIdx.y) * 16;
  const int n0   = blockIdx.y * 64;
  (void)M;

  v8f zero = {};
  v8f acc[4];
#pragma unroll
  for (int t = 0; t < 4; ++t) acc[t] = zero;

  const _Float16* arow = A + (size_t)(m0 + ln) * K;
  const _Float16* brow = Bt + (size_t)(n0 + ln) * K;

  for (int k0 = 0; k0 < K; k0 += 32) {
    __builtin_prefetch(arow + k0 + 64, 0, 1);   // global_prefetch_b8
    v16h a = load_ab(arow + k0, ko);
#pragma unroll
    for (int t = 0; t < 4; ++t) {
      v16h b = load_ab(brow + (size_t)t * 16 * K + k0, ko);
      acc[t] = __builtin_amdgcn_wmma_f32_16x16x32_f16(
          false, a, false, b, (short)0, acc[t], false, false);
    }
  }

#pragma unroll
  for (int t = 0; t < 4; ++t) {
    const int n = n0 + t * 16 + ln;
    const float bv = bias ? bias[n] : 0.0f;
#pragma unroll
    for (int j = 0; j < 8; ++j) {
      const int m = m0 + j + 8 * grp;     // C layout: row = j + 8*(lane>=16)
      const size_t idx = (size_t)m * Nn + n;
      float c = acc[t][j] + bv;
      if constexpr (MODE == 0) {
        outf[idx] = c;
      } else if constexpr (MODE == 1) {
        outh[idx] = (_Float16)c;
      } else if constexpr (MODE == 2) {
        float g = 0.5f * c * (1.0f + erff(c * 0.70710678118654752f));
        outh[idx] = (_Float16)g;
      } else if constexpr (MODE == 3) {
        outf[idx] = 1.0f / (1.0f + expf(-c));
      } else {
        outf[idx] = c + add[idx];
      }
    }
  }
}

// ---------------------------------------------------------------------------
// Flash-style masked attention. One wave = one (head, 16-row Q tile).
// Streams keys in 32-wide blocks, online softmax, never materializes scores.
// ---------------------------------------------------------------------------
__global__ void __launch_bounds__(128)
attn_wmma(const _Float16* __restrict__ qh, const _Float16* __restrict__ kh,
          const _Float16* __restrict__ vT, const unsigned char* __restrict__ keep,
          _Float16* __restrict__ ctx) {
  const int lane = threadIdx.x;
  const int w    = threadIdx.y;
  const int grp  = lane >> 4;
  const int ln   = lane & 15;
  const int ko   = grp * 8;
  const int h    = blockIdx.y;
  const int q0   = (blockIdx.x * blockDim.y + w) * 16;

  __shared__ __align__(16) _Float16 Pl[4][16][32];

  const v16h a = load_ab(qh + (size_t)(q0 + ln) * DMODEL + h * HDIM, ko);
  v8f zero = {};
  v8f o0 = zero, o1 = zero;
  float mrow[8], lrow[8];
#pragma unroll
  for (int j = 0; j < 8; ++j) { mrow[j] = -1e30f; lrow[j] = 0.0f; }

  const float scale = 0.17677669529663687f;  // 1/sqrt(32)
  const unsigned char* keep_base = keep + ((size_t)h * N_TOK + q0) * N_TOK;

  for (int kb = 0; kb < N_TOK; kb += 32) {
    v16h bk0 = load_ab(kh + (size_t)(kb + ln) * DMODEL + h * HDIM, ko);
    v16h bk1 = load_ab(kh + (size_t)(kb + 16 + ln) * DMODEL + h * HDIM, ko);
    v8f s0 = __builtin_amdgcn_wmma_f32_16x16x32_f16(false, a, false, bk0,
                                                    (short)0, zero, false, false);
    v8f s1 = __builtin_amdgcn_wmma_f32_16x16x32_f16(false, a, false, bk1,
                                                    (short)0, zero, false, false);
#pragma unroll
    for (int j = 0; j < 8; ++j) {
      const int m = j + 8 * grp;
      const unsigned char* kr = keep_base + (size_t)m * N_TOK + kb;
      float x0 = kr[ln]      ? s0[j] * scale : -1e30f;
      float x1 = kr[16 + ln] ? s1[j] * scale : -1e30f;
      // row max over the 32 columns of this block (16-lane group reduce)
      float t = fmaxf(x0, x1);
      t = fmaxf(t, __shfl_xor(t, 1, 32));
      t = fmaxf(t, __shfl_xor(t, 2, 32));
      t = fmaxf(t, __shfl_xor(t, 4, 32));
      t = fmaxf(t, __shfl_xor(t, 8, 32));
      float mn = fmaxf(mrow[j], t);
      float f  = expf(mrow[j] - mn);
      float p0 = expf(x0 - mn);
      float p1 = expf(x1 - mn);
      float rs = p0 + p1;
      rs += __shfl_xor(rs, 1, 32);
      rs += __shfl_xor(rs, 2, 32);
      rs += __shfl_xor(rs, 4, 32);
      rs += __shfl_xor(rs, 8, 32);
      lrow[j] = lrow[j] * f + rs;
      mrow[j] = mn;
      o0[j] *= f;
      o1[j] *= f;
      Pl[w][m][ln]      = (_Float16)p0;   // C layout -> row-major in LDS
      Pl[w][m][16 + ln] = (_Float16)p1;
    }
    __asm volatile("s_wait_dscnt 0" ::: "memory");
    v16h ap  = load_ab(&Pl[w][ln][0], ko);          // reload as A layout
    v16h bv0 = load_ab(vT + ((size_t)h * HDIM + ln) * N_TOK + kb, ko);
    v16h bv1 = load_ab(vT + ((size_t)h * HDIM + 16 + ln) * N_TOK + kb, ko);
    o0 = __builtin_amdgcn_wmma_f32_16x16x32_f16(false, ap, false, bv0,
                                                (short)0, o0, false, false);
    o1 = __builtin_amdgcn_wmma_f32_16x16x32_f16(false, ap, false, bv1,
                                                (short)0, o1, false, false);
  }

#pragma unroll
  for (int j = 0; j < 8; ++j) {
    const int m = q0 + j + 8 * grp;
    const float inv = lrow[j] > 0.0f ? 1.0f / lrow[j] : 0.0f;
    ctx[(size_t)m * DMODEL + h * HDIM + ln]      = (_Float16)(o0[j] * inv);
    ctx[(size_t)m * DMODEL + h * HDIM + 16 + ln] = (_Float16)(o1[j] * inv);
  }
}

// ---------------------------------------------------------------------------
// LayerNorm, one wave per row (8 elems/lane, full-wave shuffle reduction).
// Optionally writes f32 and/or f16 (into concat buffer at column offset).
// ---------------------------------------------------------------------------
__global__ void __launch_bounds__(32)
ln_kernel(const float* __restrict__ in, const float* __restrict__ g,
          const float* __restrict__ b, float* __restrict__ outf,
          _Float16* __restrict__ outh, int ldh, int offh) {
  const int n = blockIdx.x;
  const int lane = threadIdx.x;
  float v[8];
  float s = 0.0f;
#pragma unroll
  for (int i = 0; i < 8; ++i) {
    v[i] = in[(size_t)n * DMODEL + i * 32 + lane];
    s += v[i];
  }
  const float mean = wave_sum32(s) * (1.0f / DMODEL);
  float var = 0.0f;
#pragma unroll
  for (int i = 0; i < 8; ++i) { float d = v[i] - mean; var += d * d; }
  var = wave_sum32(var) * (1.0f / DMODEL);
  const float rstd = rsqrtf(var + 1e-5f);
#pragma unroll
  for (int i = 0; i < 8; ++i) {
    const int d = i * 32 + lane;
    const float y = (v[i] - mean) * rstd * g[d] + b[d];
    if (outf) outf[(size_t)n * DMODEL + d] = y;
    if (outh) outh[(size_t)n * ldh + offh + d] = (_Float16)y;
  }
}

// local = sum_r where(cnt>0, seg/cnt, 0) + x@ws ; then LN1; f16 copy -> cat[:,0:256]
__global__ void __launch_bounds__(32)
ln1_kernel(const float* __restrict__ xw, const float* __restrict__ seg,
           const float* __restrict__ cnt, const float* __restrict__ g,
           const float* __restrict__ b, float* __restrict__ outf,
           _Float16* __restrict__ cath) {
  const int n = blockIdx.x;
  const int lane = threadIdx.x;
  float inv[RREL];
#pragma unroll
  for (int r = 0; r < RREL; ++r) {
    float c = cnt[r * N_TOK + n];
    inv[r] = c > 0.0f ? 1.0f / c : 0.0f;
  }
  float v[8];
  float s = 0.0f;
#pragma unroll
  for (int i = 0; i < 8; ++i) {
    const int d = i * 32 + lane;
    float a = xw[(size_t)n * DMODEL + d];
#pragma unroll
    for (int r = 0; r < RREL; ++r)
      a += seg[((size_t)r * N_TOK + n) * DMODEL + d] * inv[r];
    v[i] = a;
    s += a;
  }
  const float mean = wave_sum32(s) * (1.0f / DMODEL);
  float var = 0.0f;
#pragma unroll
  for (int i = 0; i < 8; ++i) { float d = v[i] - mean; var += d * d; }
  var = wave_sum32(var) * (1.0f / DMODEL);
  const float rstd = rsqrtf(var + 1e-5f);
#pragma unroll
  for (int i = 0; i < 8; ++i) {
    const int d = i * 32 + lane;
    const float y = (v[i] - mean) * rstd * g[d] + b[d];
    outf[(size_t)n * DMODEL + d] = y;
    cath[(size_t)n * CATD + d] = (_Float16)y;
  }
}

// One wave per edge: seg[r][dst] += rel_feat[r][src]; cnt[r][dst] += 1
__global__ void __launch_bounds__(256)
scatter_kernel(const float* __restrict__ rel_feat, const int* __restrict__ src,
               const int* __restrict__ dst, float* __restrict__ seg,
               float* __restrict__ cnt) {
  const int wid = (int)((blockIdx.x * blockDim.x + threadIdx.x) >> 5);
  const int lane = threadIdx.x & 31;
  if (wid >= RREL * NEDGE) return;
  const int r = wid / NEDGE;
  const int e = wid - r * NEDGE;
  const int sv = src[(size_t)r * NEDGE + e];
  const int dv = dst[(size_t)r * NEDGE + e];
  const float* mrow = rel_feat + ((size_t)r * N_TOK + sv) * DMODEL;
  float* srow = seg + ((size_t)r * N_TOK + dv) * DMODEL;
#pragma unroll
  for (int i = 0; i < 8; ++i)
    atomicAdd(&srow[i * 32 + lane], mrow[i * 32 + lane]);
  if (lane == 0) atomicAdd(&cnt[(size_t)r * N_TOK + dv], 1.0f);
}

__global__ void __launch_bounds__(256)
f32_to_f16(const float* __restrict__ a, _Float16* __restrict__ o, int n) {
  const int i = blockIdx.x * blockDim.x + threadIdx.x;
  if (i < n) o[i] = (_Float16)a[i];
}

// src f32 [K][Nc] row-major  ->  dst f16 [Nc][K]  (B^T for WMMA B operand)
__global__ void __launch_bounds__(256)
transpose_to_h(const float* __restrict__ src, _Float16* __restrict__ dst,
               int K, int Nc) {
  const int idx = blockIdx.x * blockDim.x + threadIdx.x;
  if (idx >= K * Nc) return;
  const int n = idx / K;
  const int k = idx - n * K;
  dst[idx] = (_Float16)src[(size_t)k * Nc + n];
}

// relwT[r][o][i] = sum_b wc[r][b] * wb[b][i][o]   (f16, ready as GEMM B^T)
__global__ void __launch_bounds__(256)
relw_kernel(const float* __restrict__ wc, const float* __restrict__ wb,
            _Float16* __restrict__ relwT) {
  const int idx = blockIdx.x * blockDim.x + threadIdx.x;
  if (idx >= RREL * DMODEL * DMODEL) return;
  const int r = idx / (DMODEL * DMODEL);
  const int rem = idx - r * DMODEL * DMODEL;
  const int o = rem / DMODEL;
  const int i = rem - o * DMODEL;
  float s = 0.0f;
#pragma unroll
  for (int b = 0; b < NBASE; ++b)
    s += wc[r * NBASE + b] * wb[((size_t)b * DMODEL + i) * DMODEL + o];
  relwT[idx] = (_Float16)s;
}

// vT[h*HDIM+d][n] = v[n][h*HDIM+d]
__global__ void __launch_bounds__(256)
transpose_v(const _Float16* __restrict__ vh, _Float16* __restrict__ vT) {
  const int idx = blockIdx.x * blockDim.x + threadIdx.x;
  if (idx >= DMODEL * N_TOK) return;
  const int hd = idx / N_TOK;
  const int n = idx - hd * N_TOK;
  vT[idx] = vh[(size_t)n * DMODEL + hd];
}

// h = x + g*local + (1-g)*glob ; keep f32 (residual) + f16 (A for FFN)
__global__ void __launch_bounds__(256)
combine_kernel(const float* __restrict__ x, const float* __restrict__ gate,
               const float* __restrict__ loc, const float* __restrict__ glob,
               float* __restrict__ hbuf, _Float16* __restrict__ hh) {
  const int i = blockIdx.x * blockDim.x + threadIdx.x;
  if (i >= N_TOK * DMODEL) return;
  const float g = gate[i];
  const float hv = x[i] + g * loc[i] + (1.0f - g) * glob[i];
  hbuf[i] = hv;
  hh[i] = (_Float16)hv;
}

extern "C" void kernel_launch(void* const* d_in, const int* in_sizes, int n_in,
                              void* d_out, int out_size, void* d_ws, size_t ws_size,
                              hipStream_t stream) {
  (void)in_sizes; (void)n_in; (void)out_size; (void)ws_size;
  const float* x    = (const float*)d_in[0];
  const float* wb   = (const float*)d_in[1];
  const float* wc   = (const float*)d_in[2];
  const float* wsm  = (const float*)d_in[3];
  const float* ln1g = (const float*)d_in[4];
  const float* ln1b = (const float*)d_in[5];
  const float* qw   = (const float*)d_in[6];
  const float* qb   = (const float*)d_in[7];
  const float* kw   = (const float*)d_in[8];
  const float* kbp  = (const float*)d_in[9];
  const float* vw   = (const float*)d_in[10];
  const float* vbp  = (const float*)d_in[11];
  const float* ow   = (const float*)d_in[12];
  const float* ob   = (const float*)d_in[13];
  /* d_in[14] = vt, unused in forward */
  const float* ln2g = (const float*)d_in[15];
  const float* ln2b = (const float*)d_in[16];
  const float* gw   = (const float*)d_in[17];
  const float* gb   = (const float*)d_in[18];
  const float* f1w  = (const float*)d_in[19];
  const float* f1b  = (const float*)d_in[20];
  const float* f2w  = (const float*)d_in[21];
  const float* f2b  = (const float*)d_in[22];
  const float* ln3g = (const float*)d_in[23];
  const float* ln3b = (const float*)d_in[24];
  const int* src    = (const int*)d_in[25];
  const int* dst    = (const int*)d_in[26];
  const unsigned char* keep = (const unsigned char*)d_in[27];

  char* wsp = (char*)d_ws;
  size_t off = 0;
  auto alloc = [&](size_t bytes) -> void* {
    void* p = wsp + off;
    off += (bytes + 255) & ~(size_t)255;
    return p;
  };
  _Float16* x_h    = (_Float16*)alloc((size_t)N_TOK * DMODEL * 2);
  _Float16* wsT    = (_Float16*)alloc((size_t)DMODEL * DMODEL * 2);
  _Float16* qwT    = (_Float16*)alloc((size_t)DMODEL * DMODEL * 2);
  _Float16* kwT    = (_Float16*)alloc((size_t)DMODEL * DMODEL * 2);
  _Float16* vwT    = (_Float16*)alloc((size_t)DMODEL * DMODEL * 2);
  _Float16* owT    = (_Float16*)alloc((size_t)DMODEL * DMODEL * 2);
  _Float16* gwT    = (_Float16*)alloc((size_t)CATD * DMODEL * 2);
  _Float16* f1wT   = (_Float16*)alloc((size_t)DMODEL * DFF * 2);
  _Float16* f2wT   = (_Float16*)alloc((size_t)DFF * DMODEL * 2);
  _Float16* relwT  = (_Float16*)alloc((size_t)RREL * DMODEL * DMODEL * 2);
  float* rel_feat  = (float*)alloc((size_t)RREL * N_TOK * DMODEL * 4);
  float* seg       = (float*)alloc((size_t)RREL * N_TOK * DMODEL * 4);
  float* cnt       = (float*)alloc((size_t)RREL * N_TOK * 4);
  float* xw        = (float*)alloc((size_t)N_TOK * DMODEL * 4);
  float* local32   = (float*)alloc((size_t)N_TOK * DMODEL * 4);
  _Float16* q_h    = (_Float16*)alloc((size_t)N_TOK * DMODEL * 2);
  _Float16* k_h    = (_Float16*)alloc((size_t)N_TOK * DMODEL * 2);
  _Float16* v_h    = (_Float16*)alloc((size_t)N_TOK * DMODEL * 2);
  _Float16* vT_h   = (_Float16*)alloc((size_t)DMODEL * N_TOK * 2);
  _Float16* ctx_h  = (_Float16*)alloc((size_t)N_TOK * DMODEL * 2);
  float* glob_raw  = (float*)alloc((size_t)N_TOK * DMODEL * 4);
  float* glob32    = (float*)alloc((size_t)N_TOK * DMODEL * 4);
  _Float16* cat_h  = (_Float16*)alloc((size_t)N_TOK * CATD * 2);
  float* gate      = (float*)alloc((size_t)N_TOK * DMODEL * 4);
  float* hbuf      = (float*)alloc((size_t)N_TOK * DMODEL * 4);
  _Float16* h_h    = (_Float16*)alloc((size_t)N_TOK * DMODEL * 2);
  _Float16* act_h  = (_Float16*)alloc((size_t)N_TOK * DFF * 2);
  float* h2        = (float*)alloc((size_t)N_TOK * DMODEL * 4);

  const dim3 gblk(32, 4);

  // ---- input conversion + weight transposes (f32 -> f16 B^T) ----
  f32_to_f16<<<(N_TOK * DMODEL + 255) / 256, 256, 0, stream>>>(x, x_h, N_TOK * DMODEL);
  transpose_to_h<<<(DMODEL * DMODEL) / 256, 256, 0, stream>>>(wsm, wsT, DMODEL, DMODEL);
  transpose_to_h<<<(DMODEL * DMODEL) / 256, 256, 0, stream>>>(qw, qwT, DMODEL, DMODEL);
  transpose_to_h<<<(DMODEL * DMODEL) / 256, 256, 0, stream>>>(kw, kwT, DMODEL, DMODEL);
  transpose_to_h<<<(DMODEL * DMODEL) / 256, 256, 0, stream>>>(vw, vwT, DMODEL, DMODEL);
  transpose_to_h<<<(DMODEL * DMODEL) / 256, 256, 0, stream>>>(ow, owT, DMODEL, DMODEL);
  transpose_to_h<<<(CATD * DMODEL) / 256, 256, 0, stream>>>(gw, gwT, CATD, DMODEL);
  transpose_to_h<<<(DMODEL * DFF) / 256, 256, 0, stream>>>(f1w, f1wT, DMODEL, DFF);
  transpose_to_h<<<(DFF * DMODEL) / 256, 256, 0, stream>>>(f2w, f2wT, DFF, DMODEL);
  relw_kernel<<<(RREL * DMODEL * DMODEL) / 256, 256, 0, stream>>>(wc, wb, relwT);

  // ---- message passing branch ----
  gemm_wmma<0><<<dim3(32, 4), gblk, 0, stream>>>(x_h, wsT, nullptr, nullptr,
                                                 xw, nullptr, N_TOK, DMODEL, DMODEL);
  for (int r = 0; r < RREL; ++r) {
    gemm_wmma<0><<<dim3(32, 4), gblk, 0, stream>>>(
        x_h, relwT + (size_t)r * DMODEL * DMODEL, nullptr, nullptr,
        rel_feat + (size_t)r * N_TOK * DMODEL, nullptr, N_TOK, DMODEL, DMODEL);
  }
  hipMemsetAsync(seg, 0, (size_t)RREL * N_TOK * DMODEL * 4, stream);
  hipMemsetAsync(cnt, 0, (size_t)RREL * N_TOK * 4, stream);
  scatter_kernel<<<(RREL * NEDGE) / 8, 256, 0, stream>>>(rel_feat, src, dst, seg, cnt);
  ln1_kernel<<<N_TOK, 32, 0, stream>>>(xw, seg, cnt, ln1g, ln1b, local32, cat_h);

  // ---- attention branch ----
  gemm_wmma<1><<<dim3(32, 4), gblk, 0, stream>>>(x_h, qwT, qb, nullptr,
                                                 nullptr, q_h, N_TOK, DMODEL, DMODEL);
  gemm_wmma<1><<<dim3(32, 4), gblk, 0, stream>>>(x_h, kwT, kbp, nullptr,
                                                 nullptr, k_h, N_TOK, DMODEL, DMODEL);
  gemm_wmma<1><<<dim3(32, 4), gblk, 0, stream>>>(x_h, vwT, vbp, nullptr,
                                                 nullptr, v_h, N_TOK, DMODEL, DMODEL);
  transpose_v<<<(DMODEL * N_TOK) / 256, 256, 0, stream>>>(v_h, vT_h);
  attn_wmma<<<dim3(N_TOK / 64, NHEAD), gblk, 0, stream>>>(q_h, k_h, vT_h, keep, ctx_h);
  gemm_wmma<0><<<dim3(32, 4), gblk, 0, stream>>>(ctx_h, owT, ob, nullptr,
                                                 glob_raw, nullptr, N_TOK, DMODEL, DMODEL);
  ln_kernel<<<N_TOK, 32, 0, stream>>>(glob_raw, ln2g, ln2b, glob32, cat_h, CATD, DMODEL);

  // ---- gated fusion + FFN ----
  gemm_wmma<3><<<dim3(32, 4), gblk, 0, stream>>>(cat_h, gwT, gb, nullptr,
                                                 gate, nullptr, N_TOK, DMODEL, CATD);
  combine_kernel<<<(N_TOK * DMODEL + 255) / 256, 256, 0, stream>>>(
      x, gate, local32, glob32, hbuf, h_h);
  gemm_wmma<2><<<dim3(32, 16), gblk, 0, stream>>>(h_h, f1wT, f1b, nullptr,
                                                  nullptr, act_h, N_TOK, DFF, DMODEL);
  gemm_wmma<4><<<dim3(32, 4), gblk, 0, stream>>>(act_h, f2wT, f2b, hbuf,
                                                 h2, nullptr, N_TOK, DMODEL, DFF);
  ln_kernel<<<N_TOK, 32, 0, stream>>>(h2, ln3g, ln3b, (float*)d_out, nullptr, 0, 0);
}
